// SGAT_16655883174582
// MI455X (gfx1250) — compile-verified, matrix operand order checked by reading, not run on hardware
//
#include <hip/hip_runtime.h>
#include <hip/hip_bf16.h>

typedef __attribute__((ext_vector_type(16))) __bf16 v16bf;
typedef __attribute__((ext_vector_type(8)))  float  v8f;

#define Dm 128
#define Hm 64
#define Om 128
#define NFRAG 16   // 16 output-column tiles: 0-3 -> q, 4-7 -> k, 8-15 -> v
#define KSTEPS 4   // 128 / 32

// ---------------- order-preserving float<->uint keys for atomic max -------
__device__ __forceinline__ unsigned orderKey(float f) {
    unsigned u = __float_as_uint(f);
    return (u & 0x80000000u) ? ~u : (u | 0x80000000u);
}
__device__ __forceinline__ float fromKey(unsigned u) {
    return __uint_as_float((u & 0x80000000u) ? (u & 0x7FFFFFFFu) : ~u);
}

// ---------------- BatchNorm: partial column sums --------------------------
__global__ __launch_bounds__(128) void bn_partial(const float* __restrict__ feat,
                                                  float* __restrict__ colsum,
                                                  float* __restrict__ colsumsq,
                                                  int N) {
    const int c = threadIdx.x;   // 128 threads = one column each
    float s = 0.f, s2 = 0.f;
    for (int r = blockIdx.x; r < N; r += gridDim.x) {
        float f = feat[(size_t)r * Dm + c];
        s += f; s2 += f * f;
    }
    unsafeAtomicAdd(colsum + c, s);
    unsafeAtomicAdd(colsumsq + c, s2);
}

// ---------------- BatchNorm: fold into per-column scale/shift -------------
__global__ __launch_bounds__(128) void bn_finalize(const float* __restrict__ colsum,
                                                   const float* __restrict__ colsumsq,
                                                   const float* __restrict__ gamma,
                                                   const float* __restrict__ beta,
                                                   float* __restrict__ scale,
                                                   float* __restrict__ shift,
                                                   int N) {
    const int c = threadIdx.x;
    const float invN = 1.0f / (float)N;
    float mean = colsum[c] * invN;
    float var  = colsumsq[c] * invN - mean * mean;
    float rstd = rsqrtf(var + 1e-5f);
    float sc = gamma[c] * rstd;
    scale[c] = sc;
    shift[c] = beta[c] - mean * sc;
}

// ---------------- fused normalize + q/k/v projection via WMMA -------------
__global__ __launch_bounds__(128) void proj_wmma(const float* __restrict__ feat,
                                                 const float* __restrict__ Wq,
                                                 const float* __restrict__ bq,
                                                 const float* __restrict__ Wk,
                                                 const float* __restrict__ Wv,
                                                 const float* __restrict__ scale,
                                                 const float* __restrict__ shift,
                                                 float* __restrict__ q,
                                                 float* __restrict__ k,
                                                 float* __restrict__ v,
                                                 int N) {
    // Stage all weights in LDS, bf16, pre-swizzled into WMMA B-fragment order:
    // fragment f = ntile*4 + kstep; per fragment 32 lanes x 16 bf16 contiguous.
    __shared__ __bf16 wlds[NFRAG * KSTEPS * 32 * 16];   // 32768 bf16 = 64 KB
    const int tid = threadIdx.x;
    for (int idx = tid; idx < NFRAG * KSTEPS * 32 * 16; idx += 128) {
        int e    = idx & 15;
        int lane = (idx >> 4) & 31;
        int f    = idx >> 9;
        int kstep = f & 3;
        int ntile = f >> 2;
        int n = lane & 15, h = lane >> 4;
        // 16-bit B-fragment element -> K index (mirrors 16-bit A 16x32 layout)
        int kk = kstep * 32 + ((e < 8) ? (h * 8 + e) : (16 + h * 8 + (e - 8)));
        int col = ntile * 16 + n;           // global packed output column 0..255
        float wv;
        if (col < Hm)            wv = Wq[(size_t)kk * Hm + col];
        else if (col < 2 * Hm)   wv = Wk[(size_t)kk * Hm + (col - Hm)];
        else                     wv = Wv[(size_t)kk * Om + (col - 2 * Hm)];
        wlds[idx] = (__bf16)wv;
    }
    __syncthreads();

    const int lane = tid & 31;
    const int m = lane & 15, h = lane >> 4;
    const int nTiles = (N + 15) >> 4;
    const int wavesTotal = gridDim.x * 4;
    for (int t = blockIdx.x * 4 + (tid >> 5); t < nTiles; t += wavesTotal) {
        const int row0 = t << 4;
        int row = row0 + m; if (row >= N) row = N - 1;   // clamp; stores guarded
        const float* fr = feat + (size_t)row * Dm;

        // Build A fragments: normalized feat, documented 16-bit 16x32 layout.
        v16bf afrag[KSTEPS];
        #pragma unroll
        for (int ks = 0; ks < KSTEPS; ++ks) {
            const int kb = ks * 32 + h * 8;
            #pragma unroll
            for (int e = 0; e < 8; ++e) {
                int c0 = kb + e;
                int c1 = kb + 16 + e;
                afrag[ks][e]     = (__bf16)(fr[c0] * scale[c0] + shift[c0]);
                afrag[ks][e + 8] = (__bf16)(fr[c1] * scale[c1] + shift[c1]);
            }
        }

        #pragma unroll
        for (int ntile = 0; ntile < NFRAG; ++ntile) {
            v8f acc = {};
            #pragma unroll
            for (int ks = 0; ks < KSTEPS; ++ks) {
                v16bf bfrag = *(const v16bf*)&wlds[((ntile * 4 + ks) * 32 + lane) * 16];
                acc = __builtin_amdgcn_wmma_f32_16x16x32_bf16(
                          false, afrag[ks], false, bfrag,
                          (short)0, acc, false, false);
            }
            const int col = ntile * 16 + m;          // C/D: N = lane&15
            float bias; float* outp; int ocol, ostride;
            if (ntile < 4)      { outp = q; ocol = col;          ostride = Hm; bias = bq[col]; }
            else if (ntile < 8) { outp = k; ocol = col - Hm;     ostride = Hm; bias = 0.f; }
            else                { outp = v; ocol = col - 2 * Hm; ostride = Om; bias = 0.f; }
            #pragma unroll
            for (int r = 0; r < 8; ++r) {            // C/D: VGPR r -> row r + 8*h
                int orow = row0 + r + 8 * h;
                if (orow < N) outp[(size_t)orow * ostride + ocol] = acc[r] + bias;
            }
        }
    }
}

// ---------------- per-edge score + running segment max --------------------
__global__ __launch_bounds__(256) void edge_score(const float* __restrict__ q,
                                                  const float* __restrict__ k,
                                                  const float* __restrict__ we,
                                                  const int* __restrict__ src,
                                                  const int* __restrict__ dst,
                                                  float* __restrict__ esc,
                                                  unsigned* __restrict__ maxkey,
                                                  int E) {
    int e = blockIdx.x * 256 + threadIdx.x;
    if (e >= E) return;
    const int s = src[e], d = dst[e];
    const float4* qa = (const float4*)(q + (size_t)s * Hm);
    const float4* ka = (const float4*)(k + (size_t)d * Hm);
    const float4* wa = (const float4*)we;
    float acc = 0.f;
    #pragma unroll
    for (int i = 0; i < Hm / 4; ++i) {
        float4 a = qa[i], b = ka[i], w = wa[i];
        acc += w.x / (1.f + __expf(-(a.x + b.x)));
        acc += w.y / (1.f + __expf(-(a.y + b.y)));
        acc += w.z / (1.f + __expf(-(a.z + b.z)));
        acc += w.w / (1.f + __expf(-(a.w + b.w)));
    }
    esc[e] = acc;
    atomicMax(maxkey + d, orderKey(acc));
}

// ---------------- exp(e - max[dst]) and segment denom ---------------------
__global__ __launch_bounds__(256) void edge_exp(float* __restrict__ esc,
                                                const int* __restrict__ dst,
                                                const unsigned* __restrict__ maxkey,
                                                float* __restrict__ denom,
                                                int E) {
    int e = blockIdx.x * 256 + threadIdx.x;
    if (e >= E) return;
    const int d = dst[e];
    float ex = __expf(esc[e] - fromKey(maxkey[d]));
    esc[e] = ex;
    unsafeAtomicAdd(denom + d, ex);
}

// ---------------- weighted scatter-sum aggregation (wave per edge) --------
__global__ __launch_bounds__(256) void aggregate(const float* __restrict__ v,
                                                 const float* __restrict__ esc,
                                                 const float* __restrict__ denom,
                                                 const int* __restrict__ src,
                                                 const int* __restrict__ dst,
                                                 float* __restrict__ out,
                                                 int E) {
    int e = blockIdx.x * 8 + (threadIdx.x >> 5);   // one wave32 per edge
    if (e >= E) return;
    const int lane = threadIdx.x & 31;
    const int s = src[e], d = dst[e];
    const float a = esc[e] / denom[d];
    float4 vv = *(const float4*)(v + (size_t)s * Om + lane * 4);
    float* op = out + (size_t)d * Om + lane * 4;
    unsafeAtomicAdd(op + 0, vv.x * a);
    unsafeAtomicAdd(op + 1, vv.y * a);
    unsafeAtomicAdd(op + 2, vv.z * a);
    unsafeAtomicAdd(op + 3, vv.w * a);
}

extern "C" void kernel_launch(void* const* d_in, const int* in_sizes, int n_in,
                              void* d_out, int out_size, void* d_ws, size_t ws_size,
                              hipStream_t stream) {
    const float* feat  = (const float*)d_in[0];
    const float* gamma = (const float*)d_in[1];
    const float* beta  = (const float*)d_in[2];
    const float* Wq    = (const float*)d_in[3];
    const float* bq    = (const float*)d_in[4];
    const float* Wk    = (const float*)d_in[5];
    const float* Wv    = (const float*)d_in[6];
    const float* we    = (const float*)d_in[7];
    const int*   src   = (const int*)d_in[8];
    const int*   dst   = (const int*)d_in[9];
    float* out = (float*)d_out;

    const int N = in_sizes[0] / Dm;
    const int E = in_sizes[8];

    // workspace layout (floats unless noted)
    float*    q        = (float*)d_ws;
    float*    k        = q + (size_t)N * Hm;
    float*    v        = k + (size_t)N * Hm;
    float*    esc      = v + (size_t)N * Om;
    unsigned* maxkey   = (unsigned*)(esc + (size_t)E);     // zero == -inf key
    float*    denom    = (float*)(maxkey + (size_t)N);
    float*    colsum   = denom + (size_t)N;
    float*    colsumsq = colsum + Dm;
    float*    scale    = colsumsq + Dm;
    float*    shift    = scale + Dm;

    hipMemsetAsync(colsum, 0, 2 * Dm * sizeof(float), stream);          // sums
    hipMemsetAsync(maxkey, 0, 2 * (size_t)N * sizeof(float), stream);   // maxkey+denom
    hipMemsetAsync(out, 0, (size_t)N * Om * sizeof(float), stream);     // accum target

    bn_partial<<<512, 128, 0, stream>>>(feat, colsum, colsumsq, N);
    bn_finalize<<<1, 128, 0, stream>>>(colsum, colsumsq, gamma, beta, scale, shift, N);

    const int nTiles = (N + 15) / 16;
    const int projBlocks = (nTiles + 3) / 4;                 // 4 waves per block
    proj_wmma<<<projBlocks, 128, 0, stream>>>(feat, Wq, bq, Wk, Wv,
                                              scale, shift, q, k, v, N);

    edge_score<<<(E + 255) / 256, 256, 0, stream>>>(q, k, we, src, dst, esc, maxkey, E);
    edge_exp<<<(E + 255) / 256, 256, 0, stream>>>(esc, dst, maxkey, denom, E);
    aggregate<<<(E + 7) / 8, 256, 0, stream>>>(v, esc, denom, src, dst, out, E);
}
